// CutBayesFlow_2044404433164
// MI455X (gfx1250) — matrix operand order repaired and checked by reading, not compiled, not run on hardware
//
#include <hip/hip_runtime.h>
#include <hip/hip_bf16.h>
#include <math.h>

typedef __attribute__((ext_vector_type(16))) __bf16 v16bf;
typedef __attribute__((ext_vector_type(8)))  __bf16 v8bf;
typedef __attribute__((ext_vector_type(8)))  float  v8f;

#define F_DIMC 64
#define HIDC   256
#define PETA   32
#define MULTC  47
#define NLAY   8
#define OUTC   (F_DIMC*MULTC)   /* 3008 */
#define NB     8192
#define MD     256
#define TDIM   65

// ---------- helpers ----------
static __device__ inline __bf16 f2bf(float f){
  unsigned u = __builtin_bit_cast(unsigned, f);
  unsigned r = u + 0x7FFFu + ((u >> 16) & 1u);
  unsigned short s = (unsigned short)(r >> 16);
  return __builtin_bit_cast(__bf16, s);
}

static __device__ inline float softplusf(float v){
  return v > 20.f ? v : log1pf(expf(v));
}

// A fragment (16x32 bf16) from pre-packed bf16 row-major matrix.
// lane&15 = M row, hi = lane>>4: element i <-> K = (i<8? i : i+8) + hi*8 (+k0)
static __device__ inline v16bf load_a_bf(const __bf16* __restrict__ A, int ld,
                                         int m, int k0, int lane){
  int hi = lane >> 4;
  const __bf16* p = A + m*ld + k0 + hi*8;
  v8bf c0 = *(const v8bf*)(p);        // K chunk [k0+hi*8, +8)
  v8bf c1 = *(const v8bf*)(p + 16);   // K chunk [k0+16+hi*8, +8)
  return __builtin_shufflevector(c0, c1, 0,1,2,3,4,5,6,7,8,9,10,11,12,13,14,15);
}

// B fragment (32x16 bf16): weight kept [N][K] row-major bf16 (=> out = A @ W^T).
static __device__ inline v16bf load_b_bf(const __bf16* __restrict__ W, int ld,
                                         int n, int k0, int lane){
  int hi = lane >> 4;
  const __bf16* p = W + n*ld + k0 + hi*8;
  v8bf c0 = *(const v8bf*)(p);
  v8bf c1 = *(const v8bf*)(p + 16);
  return __builtin_shufflevector(c0, c1, 0,1,2,3,4,5,6,7,8,9,10,11,12,13,14,15);
}

static __device__ inline v8f wmma_bf16(v16bf a, v16bf b, v8f c){
  return __builtin_amdgcn_wmma_f32_16x16x32_bf16(false, a, false, b, (short)0, c,
                                                 false, false);
}

// ---------- weight prep (mask + f32->bf16) ----------
__global__ void prep_wi(const float* __restrict__ W, __bf16* __restrict__ O){
  int idx = blockIdx.x*256 + threadIdx.x;
  if (idx >= NLAY*HIDC*F_DIMC) return;
  int j = idx & 63;
  int i = (idx >> 6) & 255;
  float mask = ((i % 63) >= j) ? 1.f : 0.f;          // hid_deg >= in_deg
  O[idx] = f2bf(W[idx] * mask);
}
__global__ void prep_wc(const float* __restrict__ W, __bf16* __restrict__ O){
  int idx = blockIdx.x*256 + threadIdx.x;
  if (idx >= NLAY*HIDC*PETA) return;
  O[idx] = f2bf(W[idx]);
}
__global__ void prep_wb(const float* __restrict__ W, __bf16* __restrict__ O){
  int idx = blockIdx.x*256 + threadIdx.x;
  if (idx >= NLAY*4*HIDC*HIDC) return;
  int j = idx & 255;
  int i = (idx >> 8) & 255;
  float mask = ((i % 63) >= (j % 63)) ? 1.f : 0.f;   // m_hh
  O[idx] = f2bf(W[idx] * mask);
}
__global__ void prep_wo(const float* __restrict__ W, __bf16* __restrict__ O){
  int idx = blockIdx.x*256 + threadIdx.x;
  if (idx >= NLAY*OUTC*HIDC) return;
  int i = idx & 255;
  int o = (idx >> 8) % OUTC;
  float mask = ((o / MULTC) > (i % 63)) ? 1.f : 0.f; // out_deg > hid_deg
  O[idx] = f2bf(W[idx] * mask);
}

__global__ void zero_f32(float* p, int n){
  int i = blockIdx.x*256 + threadIdx.x;
  if (i < n) p[i] = 0.f;
}

__global__ void cvt_eta(const float* __restrict__ E, __bf16* __restrict__ O){
  int i = blockIdx.x*256 + threadIdx.x;
  if (i < NB*PETA) O[i] = f2bf(E[i]);
}

// flip features; emit f32 (spline input) and bf16 (GEMM A) copies
__global__ void flip_x(const float* __restrict__ src, float* __restrict__ dst,
                       __bf16* __restrict__ dbf){
  int i = blockIdx.x*256 + threadIdx.x;
  if (i >= NB*F_DIMC) return;
  int m = i >> 6, j = i & 63;
  float v = src[m*F_DIMC + (63 - j)];
  dst[m*F_DIMC + j] = v;
  dbf[m*F_DIMC + j] = f2bf(v);
}

// ---------- GEMM: H = Xf @ Wi^T + bi + Eta @ Wc^T + bc ----------
// 16x64 tile per wave; 8 waves of a block share the same 64 output columns.
__global__ void gemm_in(const __bf16* __restrict__ Xf, const __bf16* __restrict__ Eta,
                        const __bf16* __restrict__ Wi, const __bf16* __restrict__ Wc,
                        const float* __restrict__ bi, const float* __restrict__ bc,
                        float* __restrict__ H, __bf16* __restrict__ HR){
  int lane = threadIdx.x & 31, wv = threadIdx.x >> 5;
  int bm = blockIdx.x >> 2, tq = blockIdx.x & 3;   // 64 x 4 blocks
  int m0 = (bm*8 + wv)*16, n0q = tq*64;
  int nl = lane & 15, hi = lane >> 4;
  int m = m0 + nl;
  v8f acc[4];
#pragma unroll
  for (int j = 0; j < 4; j++) acc[j] = (v8f){0.f,0.f,0.f,0.f,0.f,0.f,0.f,0.f};
#pragma unroll
  for (int k0 = 0; k0 < F_DIMC; k0 += 32){
    v16bf a = load_a_bf(Xf, F_DIMC, m, k0, lane);
#pragma unroll
    for (int j = 0; j < 4; j++){
      v16bf b = load_b_bf(Wi, F_DIMC, n0q + j*16 + nl, k0, lane);
      acc[j] = wmma_bf16(a, b, acc[j]);
    }
  }
  {
    v16bf a = load_a_bf(Eta, PETA, m, 0, lane);
#pragma unroll
    for (int j = 0; j < 4; j++){
      v16bf b = load_b_bf(Wc, PETA, n0q + j*16 + nl, 0, lane);
      acc[j] = wmma_bf16(a, b, acc[j]);
    }
  }
#pragma unroll
  for (int j = 0; j < 4; j++){
    int n = n0q + j*16 + nl;
    float bv = bi[n] + bc[n];
#pragma unroll
    for (int r = 0; r < 8; r++){
      int row = m0 + r + hi*8;
      float v = acc[j][r] + bv;
      H [row*HIDC + n] = v;
      HR[row*HIDC + n] = f2bf(fmaxf(v, 0.f));     // relu'd bf16 for next GEMM
    }
  }
}

// ---------- block GEMM #1: TR = bf16(relu(A @ W^T + bias)) ----------
__global__ void gemm_blk_t(const __bf16* __restrict__ Abf, const __bf16* __restrict__ W,
                           const float* __restrict__ bias, __bf16* __restrict__ TR){
  int lane = threadIdx.x & 31, wv = threadIdx.x >> 5;
  int bm = blockIdx.x >> 2, tq = blockIdx.x & 3;
  int m0 = (bm*8 + wv)*16, n0q = tq*64;
  int nl = lane & 15, hi = lane >> 4;
  int m = m0 + nl;
  v8f acc[4];
#pragma unroll
  for (int j = 0; j < 4; j++) acc[j] = (v8f){0.f,0.f,0.f,0.f,0.f,0.f,0.f,0.f};
#pragma unroll
  for (int k0 = 0; k0 < HIDC; k0 += 32){
    v16bf a = load_a_bf(Abf, HIDC, m, k0, lane);
#pragma unroll
    for (int j = 0; j < 4; j++){
      v16bf b = load_b_bf(W, HIDC, n0q + j*16 + nl, k0, lane);
      acc[j] = wmma_bf16(a, b, acc[j]);
    }
  }
#pragma unroll
  for (int j = 0; j < 4; j++){
    int n = n0q + j*16 + nl;
    float bv = bias[n];
#pragma unroll
    for (int r = 0; r < 8; r++){
      int row = m0 + r + hi*8;
      TR[row*HIDC + n] = f2bf(fmaxf(acc[j][r] + bv, 0.f));
    }
  }
}

// ---------- block GEMM #2: H += A @ W^T + bias; emit HR=bf16(relu(H)), Hb=bf16(H)
__global__ void gemm_blk_h(const __bf16* __restrict__ Abf, const __bf16* __restrict__ W,
                           const float* __restrict__ bias, float* __restrict__ H,
                           __bf16* __restrict__ HR, __bf16* __restrict__ Hb){
  int lane = threadIdx.x & 31, wv = threadIdx.x >> 5;
  int bm = blockIdx.x >> 2, tq = blockIdx.x & 3;
  int m0 = (bm*8 + wv)*16, n0q = tq*64;
  int nl = lane & 15, hi = lane >> 4;
  int m = m0 + nl;
  v8f acc[4];
#pragma unroll
  for (int j = 0; j < 4; j++) acc[j] = (v8f){0.f,0.f,0.f,0.f,0.f,0.f,0.f,0.f};
#pragma unroll
  for (int k0 = 0; k0 < HIDC; k0 += 32){
    v16bf a = load_a_bf(Abf, HIDC, m, k0, lane);
#pragma unroll
    for (int j = 0; j < 4; j++){
      v16bf b = load_b_bf(W, HIDC, n0q + j*16 + nl, k0, lane);
      acc[j] = wmma_bf16(a, b, acc[j]);
    }
  }
#pragma unroll
  for (int j = 0; j < 4; j++){
    int n = n0q + j*16 + nl;
    float bv = bias[n];
#pragma unroll
    for (int r = 0; r < 8; r++){
      int row = m0 + r + hi*8;
      float* dst = H + row*HIDC + n;
      float v = *dst + acc[j][r] + bv;
      *dst = v;
      HR[row*HIDC + n] = f2bf(fmaxf(v, 0.f));
      Hb[row*HIDC + n] = f2bf(v);
    }
  }
}

// ---------- fused output GEMM + RQS spline ----------
// blockIdx = brow*64 + f : all 8 waves share feature f (B rows hit WGP$).
// One wave: 16 batch rows x 1 feature (47 spline params), 3 WMMA col-tiles.
__global__ void gemm_out_spline(const __bf16* __restrict__ Hb,
                                const __bf16* __restrict__ Wo,
                                const float* __restrict__ bo,
                                const float* __restrict__ Xf,
                                float* __restrict__ Xn,
                                float* __restrict__ ldf){
  __shared__ float lds[8*16*48];
  int lane = threadIdx.x & 31, wv = threadIdx.x >> 5;
  int f = blockIdx.x & 63, brow = blockIdx.x >> 6;
  int m0 = (brow*8 + wv)*16;
  int nl = lane & 15, hi = lane >> 4;
  int m = m0 + nl;
  float* myLds = lds + wv*16*48;

  // hoist A fragments for all 8 K-steps (reused by the 3 column tiles)
  v16bf afr[8];
#pragma unroll
  for (int kk = 0; kk < 8; kk++)
    afr[kk] = load_a_bf(Hb, HIDC, m, kk*32, lane);

#pragma unroll
  for (int t = 0; t < 3; t++){
    int n0 = f*MULTC + t*16;
    int n = n0 + nl; if (n > OUTC-1) n = OUTC-1;   // last-feature pad col
    v8f acc = {0.f,0.f,0.f,0.f,0.f,0.f,0.f,0.f};
#pragma unroll
    for (int kk = 0; kk < 8; kk++){
      v16bf b = load_b_bf(Wo, HIDC, n, kk*32, lane);
      acc = wmma_bf16(afr[kk], b, acc);
    }
    float bv = bo[n];
#pragma unroll
    for (int r = 0; r < 8; r++)
      myLds[(r + hi*8)*48 + t*16 + nl] = acc[r] + bv;
  }
  __syncthreads();

  if (lane < 16){
    int m2 = m0 + lane;
    float* o = myLds + lane*48;            // 47 valid params
    const float inv16 = 1.f/16.f;          // 1/sqrt(HID)
    const float AW = 1.f - 16.f*1e-8f;
    float xin = Xf[m2*F_DIMC + f];
    bool inside = (xin >= -10.f) && (xin <= 10.f);
    float xs = fminf(fmaxf(xin, -10.f), 10.f);

    float mw = -1e30f, mh = -1e30f;
#pragma unroll
    for (int i = 0; i < 16; i++){
      mw = fmaxf(mw, o[i]);
      mh = fmaxf(mh, o[16+i]);
    }
    mw *= inv16; mh *= inv16;
    float sw = 0.f, sh = 0.f;
#pragma unroll
    for (int i = 0; i < 16; i++){
      float e0 = expf(o[i]*inv16 - mw);    sw += e0; o[i]    = e0;
      float e1 = expf(o[16+i]*inv16 - mh); sh += e1; o[16+i] = e1;
    }
    float rw = AW/sw, rh = AW/sh;

    float cwk = -10.f, chk = -10.f, csw = 0.f, csh = 0.f;
    int idx = 0; bool found = false;
    float icw = -10.f, iwd = 1.f, ich = -10.f, ihd = 1.f;
#pragma unroll
    for (int k = 0; k < 16; k++){
      csw += 1e-8f + o[k]*rw;
      csh += 1e-8f + o[16+k]*rh;
      float cwn = (k == 15) ? 10.f : 20.f*csw - 10.f;
      float chn = (k == 15) ? 10.f : 20.f*csh - 10.f;
      bool pick = (!found) && ((xs < cwn) || (k == 15));
      if (pick){
        found = true; idx = k;
        icw = cwk; iwd = cwn - cwk;
        ich = chk; ihd = chn - chk;
      }
      cwk = cwn; chk = chn;
    }
    float d0 = (idx == 0)  ? 1.f : 1e-8f + softplusf(o[31 + idx]);
    float d1 = (idx == 15) ? 1.f : 1e-8f + softplusf(o[32 + idx]);
    float delta = ihd / iwd;
    float tv = (xs - icw) / iwd;
    float tt = tv * (1.f - tv);
    float denom = delta + (d0 + d1 - 2.f*delta)*tt;
    float y = ich + ihd*(delta*tv*tv + d0*tt)/denom;
    float dnum = delta*delta*(d1*tv*tv + 2.f*delta*tt + d0*(1.f-tv)*(1.f-tv));
    float ldv = logf(dnum) - 2.f*logf(denom);
    y   = inside ? y   : xin;
    ldv = inside ? ldv : 0.f;
    Xn[m2*F_DIMC + f]  = y;
    ldf[m2*F_DIMC + f] = ldv;
  }
}

__global__ void reduce_ldf(const float* __restrict__ ldf, float* __restrict__ LD){
  int m = blockIdx.x*256 + threadIdx.x;
  if (m >= NB) return;
  float s = 0.f;
#pragma unroll
  for (int j = 0; j < F_DIMC; j++) s += ldf[m*F_DIMC + j];
  LD[m] += s;
}

// ---------- data statistics: d_mean[65], d_sq ----------
__global__ void data_stats(const float* __restrict__ D, float* __restrict__ stats){
  __shared__ float red[256];
  int t = threadIdx.x;
  float s = 0.f;
  for (int i = t; i < MD*TDIM; i += 256){ float v = D[i]; s += v*v; }
  red[t] = s; __syncthreads();
  for (int off = 128; off > 0; off >>= 1){
    if (t < off) red[t] += red[t+off];
    __syncthreads();
  }
  if (t == 0) stats[TDIM] = red[0] / (float)MD;
  if (t < TDIM){
    float c = 0.f;
    for (int r = 0; r < MD; r++) c += D[r*TDIM + t];
    stats[t] = c / (float)MD;
  }
}

// ---------- stick breaking + per-row objective ----------
__global__ void final_row(const float* __restrict__ X, const float* __restrict__ Z,
                          const float* __restrict__ LD, const float* __restrict__ stats,
                          float* __restrict__ rowval){
  int m = blockIdx.x*256 + threadIdx.x;
  if (m >= NB) return;
  const float* x = X + m*F_DIMC;
  const float* z = Z + m*F_DIMC;
  float sum_lv = 0.f, sum_wl1v = 0.f, prefix = 0.f;
  float logp = 0.f, dot = 0.f, tsq = 0.f, zsq = 0.f;
  for (int j = 0; j < F_DIMC; j++){
    float xv = x[j];
    float lv  = -softplusf(-xv);
    float l1v = -softplusf(xv);
    sum_lv   += lv;
    sum_wl1v += (float)(F_DIMC - j) * l1v;
    float th = expf(lv + prefix);
    logp += logf(fmaxf(th, 1e-10f));
    dot  += th * stats[j];
    tsq  += th * th;
    prefix += l1v;
    float zv = z[j]; zsq += zv*zv;
  }
  float thl = expf(prefix);
  logp += logf(fmaxf(thl, 1e-10f));
  dot  += thl * stats[F_DIMC];
  tsq  += thl * thl;
  float sb_ld   = sum_lv + sum_wl1v;
  float log_qz  = -0.5f*zsq - 0.5f*(float)F_DIMC*logf(2.f*3.14159265358979323846f);
  float log_q   = log_qz - (LD[m] + sb_ld);
  float log_lik = -0.5f*(stats[TDIM] - 2.f*dot + tsq);
  rowval[m] = log_q - logp - log_lik;
}

__global__ void reduce_final(const float* __restrict__ rowval, float* __restrict__ out){
  __shared__ float red[256];
  int t = threadIdx.x;
  float s = 0.f;
  for (int i = t; i < NB; i += 256) s += rowval[i];
  red[t] = s; __syncthreads();
  for (int off = 128; off > 0; off >>= 1){
    if (t < off) red[t] += red[t+off];
    __syncthreads();
  }
  if (t == 0) out[0] = red[0] / (float)NB;
}

// ---------- host orchestration ----------
extern "C" void kernel_launch(void* const* d_in, const int* in_sizes, int n_in,
                              void* d_out, int out_size, void* d_ws, size_t ws_size,
                              hipStream_t stream) {
  const float* z     = (const float*)d_in[0];
  const float* eta   = (const float*)d_in[1];
  const float* dat   = (const float*)d_in[2];
  const float* W_in  = (const float*)d_in[3];
  const float* b_in  = (const float*)d_in[4];
  const float* W_ctx = (const float*)d_in[5];
  const float* b_ctx = (const float*)d_in[6];
  const float* W_blk = (const float*)d_in[7];
  const float* b_blk = (const float*)d_in[8];
  const float* W_out = (const float*)d_in[9];
  const float* b_out = (const float*)d_in[10];
  float* out = (float*)d_out;

  char* w = (char*)d_ws;
  size_t off = 0;
  auto take = [&](size_t bytes)->char*{
    char* p = w + off;
    off += (bytes + 255) & ~(size_t)255;
    return p;
  };
  __bf16* Wi_bf  = (__bf16*)take((size_t)NLAY*HIDC*F_DIMC*2);
  __bf16* Wc_bf  = (__bf16*)take((size_t)NLAY*HIDC*PETA*2);
  __bf16* Wb_bf  = (__bf16*)take((size_t)NLAY*4*HIDC*HIDC*2);
  __bf16* Wo_bf  = (__bf16*)take((size_t)NLAY*OUTC*HIDC*2);
  float*  X0     = (float*)take((size_t)NB*F_DIMC*4);
  float*  X1     = (float*)take((size_t)NB*F_DIMC*4);
  __bf16* XFbf   = (__bf16*)take((size_t)NB*F_DIMC*2);
  __bf16* ETAbf  = (__bf16*)take((size_t)NB*PETA*2);
  float*  H      = (float*)take((size_t)NB*HIDC*4);
  __bf16* HRbf   = (__bf16*)take((size_t)NB*HIDC*2);
  __bf16* TRbf   = (__bf16*)take((size_t)NB*HIDC*2);
  __bf16* Hbf    = (__bf16*)take((size_t)NB*HIDC*2);
  float*  LD     = (float*)take((size_t)NB*4);
  float*  LDF    = (float*)take((size_t)NB*F_DIMC*4);
  float*  ROWVAL = (float*)take((size_t)NB*4);
  float*  STATS  = (float*)take((size_t)(TDIM+1)*4);

  prep_wi<<<(NLAY*HIDC*F_DIMC+255)/256, 256, 0, stream>>>(W_in,  Wi_bf);
  prep_wc<<<(NLAY*HIDC*PETA +255)/256, 256, 0, stream>>>(W_ctx, Wc_bf);
  prep_wb<<<(NLAY*4*HIDC*HIDC+255)/256, 256, 0, stream>>>(W_blk, Wb_bf);
  prep_wo<<<(NLAY*OUTC*HIDC +255)/256, 256, 0, stream>>>(W_out, Wo_bf);
  zero_f32<<<(NB+255)/256, 256, 0, stream>>>(LD, NB);
  cvt_eta<<<(NB*PETA+255)/256, 256, 0, stream>>>(eta, ETAbf);
  data_stats<<<1, 256, 0, stream>>>(dat, STATS);

  const int gemm_blocks = 256;   // 64 M-groups x 4 N-quads, 8 waves/block
  for (int l = 0; l < NLAY; l++){
    const float* src = (l == 0) ? z : X0;
    flip_x<<<(NB*F_DIMC+255)/256, 256, 0, stream>>>(src, X1, XFbf);
    gemm_in<<<gemm_blocks, 256, 0, stream>>>(
        XFbf, ETAbf,
        Wi_bf + (size_t)l*HIDC*F_DIMC, Wc_bf + (size_t)l*HIDC*PETA,
        b_in + (size_t)l*HIDC, b_ctx + (size_t)l*HIDC, H, HRbf);
    for (int blk = 0; blk < 2; blk++){
      size_t q0 = ((size_t)l*2 + blk)*2 + 0;
      size_t q1 = ((size_t)l*2 + blk)*2 + 1;
      gemm_blk_t<<<gemm_blocks, 256, 0, stream>>>(
          HRbf, Wb_bf + q0*HIDC*HIDC, b_blk + q0*HIDC, TRbf);
      gemm_blk_h<<<gemm_blocks, 256, 0, stream>>>(
          TRbf, Wb_bf + q1*HIDC*HIDC, b_blk + q1*HIDC, H, HRbf, Hbf);
    }
    gemm_out_spline<<<64*64, 256, 0, stream>>>(
        Hbf, Wo_bf + (size_t)l*OUTC*HIDC, b_out + (size_t)l*OUTC, X1, X0, LDF);
    reduce_ldf<<<(NB+255)/256, 256, 0, stream>>>(LDF, LD);
  }
  final_row<<<(NB+255)/256, 256, 0, stream>>>(X0, z, LD, STATS, ROWVAL);
  reduce_final<<<1, 256, 0, stream>>>(ROWVAL, out);
}
